// AttentionBlock_60498909331705
// MI455X (gfx1250) — compile-verified, matrix operand order checked by reading, not run on hardware
//
#include <hip/hip_runtime.h>
#include <hip/hip_bf16.h>

// ---------------------------------------------------------------------------
// AttentionBlock for MI455X (gfx1250, wave32, WMMA, async-to-LDS streaming).
//   B=8, N=2048, D=512 for all of D_IN/D_HID/D_OUT.
// Pipeline (all bf16 WMMA, f32 accumulate):
//   1) cast x, Wq, Wk, Wv  f32 -> bf16
//   2) proj GEMM: qb, kb = x*Wq^T, x*Wk^T (bf16); vt = (x*Wv^T)^T (bf16)
//   3) column softmax stats: L[b,k] = logsumexp over q of (q . k)
//   4) out[b,q,o] = sum_k exp(s[b,q,k]-L[b,k]) * v[b,k,o]   (f32 output)
// Global->LDS staging uses GLOBAL_LOAD_ASYNC_TO_LDS_B128 (ASYNCcnt) with
// double-buffered LDS so DMA overlaps WMMA.  No 134MB score tensor.
// ---------------------------------------------------------------------------

typedef __attribute__((ext_vector_type(16))) __bf16 v16bf;
typedef __attribute__((ext_vector_type(8)))  __bf16 v8bf;
typedef __attribute__((ext_vector_type(8)))  float  v8f;

#ifndef __has_builtin
#define __has_builtin(x) 0
#endif

__device__ __forceinline__ v8f wmma_bf16(v16bf a, v16bf b, v8f c) {
  return __builtin_amdgcn_wmma_f32_16x16x32_bf16(
      /*neg_a=*/false, a, /*neg_b=*/false, b,
      /*c_mod=*/(short)0, c, /*reuse_a=*/false, /*reuse_b=*/false);
}

// Generic LDS pointer -> hardware LDS byte address (addr[31:0] per ISA
// aperture rules).  The empty asm blocks constant folding so the value can
// never end up in a static initializer (ld.lld can't relocate that).
__device__ __forceinline__ uint32_t lds_base_addr(void* p) {
  uint32_t a = (uint32_t)(uintptr_t)p;
  asm volatile("" : "+v"(a));
  return a;
}

// One 16B async global->LDS transfer per enabled lane (ASYNCcnt-tracked).
__device__ __forceinline__ void async_cp16(uint32_t lds, const void* g) {
  asm volatile("global_load_async_to_lds_b128 %0, %1, off"
               :: "v"(lds), "v"((unsigned long long)(uintptr_t)g)
               : "memory");
}

__device__ __forceinline__ void wait_async0() {
#if __has_builtin(__builtin_amdgcn_s_wait_asynccnt)
  __builtin_amdgcn_s_wait_asynccnt(0);
#else
  asm volatile("s_wait_asynccnt 0x0" ::: "memory");
#endif
}

// Contiguous async copy: CHUNKS x 16B, CHUNKS must be a multiple of 256.
template <int CHUNKS>
__device__ __forceinline__ void copy_async_lin(uint32_t dst, const void* src) {
#pragma unroll
  for (int it = 0; it < CHUNKS / 256; ++it) {
    const int i = it * 256 + threadIdx.x;
    async_cp16(dst + i * 16, (const char*)src + (size_t)i * 16);
  }
}

// Strided async copy: ROWS rows of 64B (4 chunks), source row stride LDB
// bytes, packed into LDS rows of 64B.  ROWS*4 must be a multiple of 256.
template <int ROWS, int LDB>
__device__ __forceinline__ void copy_async_rows64(uint32_t dst, const void* src) {
#pragma unroll
  for (int it = 0; it < ROWS * 4 / 256; ++it) {
    const int i = it * 256 + threadIdx.x;
    const int r = i >> 2, c = i & 3;
    async_cp16(dst + i * 16, (const char*)src + (size_t)r * LDB + c * 16);
  }
}

// Load one 16x32 bf16 operand tile from LDS (row-major, leading dim `ld`
// elements).  ISA A-layout: lanes 0-15 = row lane, K {0..7, 16..23};
// lanes 16-31 = row lane-16, K {8..15, 24..31}.  Every B source here is
// stored [col][K] row-major, so identical addressing yields the B operand.
__device__ __forceinline__ v16bf load_frag(const __bf16* __restrict__ base, int ld) {
  const int lane = threadIdx.x & 31;
  const int r  = lane & 15;
  const int kb = (lane >> 4) << 3;          // 0 or 8
  const __bf16* p = base + (size_t)r * ld + kb;
  union { v16bf v; v8bf h[2]; } u;
  u.h[0] = *(const v8bf*)(p);
  u.h[1] = *(const v8bf*)(p + 16);
  return u.v;
}

// ---------------------------------------------------------------------------
__global__ void cast_f32_bf16(const float* __restrict__ s, __bf16* __restrict__ d, int n) {
  int i = blockIdx.x * blockDim.x + threadIdx.x;
  const int stride = gridDim.x * blockDim.x;
  for (; i < n; i += stride) d[i] = (__bf16)s[i];
}

// ---------------------------------------------------------------------------
// O[m,n] = sum_d A[m,d] * W[n,d];  A:[16384,512] bf16, W:[512,512] bf16.
// Block tile 128x128, 8 waves = 2(M)x4(N), each wave 64x32 (8 accum tiles).
// Double-buffered 128x32 A/W panels streamed with async-to-LDS.
__global__ __launch_bounds__(256) void proj_gemm(const __bf16* __restrict__ A,
                                                 const __bf16* __restrict__ W,
                                                 __bf16* __restrict__ O,
                                                 int transposeV) {
  extern __shared__ char smem[];
  constexpr uint32_t ASH0 = 0, ASH1 = 8192, BSH0 = 16384, BSH1 = 24576;
  __bf16* ash0 = (__bf16*)(smem + ASH0);
  __bf16* ash1 = (__bf16*)(smem + ASH1);
  __bf16* bsh0 = (__bf16*)(smem + BSH0);
  __bf16* bsh1 = (__bf16*)(smem + BSH1);
  const uint32_t base = lds_base_addr(smem);

  const int m0 = blockIdx.y * 128;
  const int n0 = blockIdx.x * 128;
  const int wv  = threadIdx.x >> 5;
  const int wm0 = (wv >> 2) * 64;
  const int wn0 = (wv & 3) * 32;
  const int lane = threadIdx.x & 31;
  const int nn = lane & 15, hl = lane >> 4;

  auto stage = [&](int step, int buf) {
    const int kk = step * 32;
    copy_async_rows64<128, 1024>(base + (buf ? ASH1 : ASH0),
                                 A + (size_t)m0 * 512 + kk);
    copy_async_rows64<128, 1024>(base + (buf ? BSH1 : BSH0),
                                 W + (size_t)n0 * 512 + kk);
  };

  v8f z = {};
  v8f acc[8];
#pragma unroll
  for (int t = 0; t < 8; ++t) acc[t] = z;

  stage(0, 0);
#pragma unroll 1
  for (int s = 0; s < 16; ++s) {
    wait_async0();
    __syncthreads();
    if (s + 1 < 16) stage(s + 1, (s + 1) & 1);
    const __bf16* ab = (s & 1) ? ash1 : ash0;
    const __bf16* bb = (s & 1) ? bsh1 : bsh0;
    v16bf af[4], bf2[2];
#pragma unroll
    for (int i = 0; i < 4; ++i) af[i]  = load_frag(ab + (size_t)(wm0 + i * 16) * 32, 32);
#pragma unroll
    for (int i = 0; i < 2; ++i) bf2[i] = load_frag(bb + (size_t)(wn0 + i * 16) * 32, 32);
#pragma unroll
    for (int tm = 0; tm < 4; ++tm)
#pragma unroll
      for (int tn = 0; tn < 2; ++tn)
        acc[tm * 2 + tn] = wmma_bf16(af[tm], bf2[tn], acc[tm * 2 + tn]);
  }

#pragma unroll
  for (int tm = 0; tm < 4; ++tm)
#pragma unroll
    for (int tn = 0; tn < 2; ++tn)
#pragma unroll
      for (int j = 0; j < 8; ++j) {
        const int gm = m0 + wm0 + tm * 16 + j + 8 * hl;
        const int gn = n0 + wn0 + tn * 16 + nn;
        const __bf16 val = (__bf16)acc[tm * 2 + tn][j];
        if (!transposeV) {
          O[(size_t)gm * 512 + gn] = val;
        } else {
          const int bi = gm >> 11, row = gm & 2047;     // batch, seq pos
          O[((size_t)bi * 512 + gn) * 2048 + row] = val;
        }
      }
}

// ---------------------------------------------------------------------------
// Per-key-column logsumexp over all 2048 queries.
// Block (kblock of 128 keys, batch).  128x512 bf16 K panel resident in LDS
// (128 KB); 16x512 Q tiles double-buffered and streamed async.
__global__ __launch_bounds__(256) void col_softmax_stats(const __bf16* __restrict__ Q,
                                                         const __bf16* __restrict__ K,
                                                         float* __restrict__ L) {
  extern __shared__ char smem[];
  constexpr uint32_t KSH = 0, QSH0 = 131072, QSH1 = 147456;
  __bf16* ksh  = (__bf16*)(smem + KSH);       // 128 x 512 (128 KB)
  __bf16* qsh0 = (__bf16*)(smem + QSH0);      // 16 x 512  (16 KB)
  __bf16* qsh1 = (__bf16*)(smem + QSH1);
  const uint32_t base = lds_base_addr(smem);

  const int b   = blockIdx.y;
  const int kb0 = blockIdx.x * 128;
  const int wv  = threadIdx.x >> 5;
  const int lane = threadIdx.x & 31;

  copy_async_lin<8192>(base + KSH,  K + ((size_t)b * 2048 + kb0) * 512);
  copy_async_lin<1024>(base + QSH0, Q + ((size_t)b * 2048) * 512);

  float mx = -__builtin_inff(), sz = 0.f;
#pragma unroll 1
  for (int qt = 0; qt < 128; ++qt) {
    wait_async0();
    __syncthreads();
    if (qt + 1 < 128)
      copy_async_lin<1024>(base + (((qt + 1) & 1) ? QSH1 : QSH0),
                           Q + ((size_t)b * 2048 + (qt + 1) * 16) * 512);
    const __bf16* qb = (qt & 1) ? qsh1 : qsh0;
    v8f s = {};
#pragma unroll
    for (int ks = 0; ks < 16; ++ks) {
      v16bf a  = load_frag(qb + ks * 32, 512);
      v16bf bb = load_frag(ksh + (size_t)(wv * 16) * 512 + ks * 32, 512);
      s = wmma_bf16(a, bb, s);
    }
    // lane covers 8 rows of its column; online logsumexp update
    float tm = s[0];
#pragma unroll
    for (int j = 1; j < 8; ++j) tm = fmaxf(tm, s[j]);
    float te = 0.f;
#pragma unroll
    for (int j = 0; j < 8; ++j) te += __expf(s[j] - tm);
    const float nm = fmaxf(mx, tm);
    sz = sz * __expf(mx - nm) + te * __expf(tm - nm);
    mx = nm;
  }
  // merge the two lane-halves holding the same column (rows 0-7 vs 8-15)
  const float om = __shfl_xor(mx, 16, 32);
  const float oz = __shfl_xor(sz, 16, 32);
  const float nm = fmaxf(mx, om);
  const float zz = sz * __expf(mx - nm) + oz * __expf(om - nm);
  if (lane < 16)
    L[(size_t)b * 2048 + kb0 + wv * 16 + lane] = nm + __logf(zz);
}

// ---------------------------------------------------------------------------
// out[b,q,o] = sum_k exp(q.k - L[b,k]) * v[k,o].
// Block (qblock of 64, batch): 64x512 f32 output.  8 waves = 4(q)x2(o).
// 64KB Q panel resident; 32-key K / V^T slices double-buffered + async.
// Per step: S tile via 16 WMMA over D=512, exp -> bf16 P in LDS, then
// 16 WMMA (K=32) accumulating into out.
__global__ __launch_bounds__(256) void attn_out(const __bf16* __restrict__ Q,
                                                const __bf16* __restrict__ K,
                                                const __bf16* __restrict__ VT,
                                                const float* __restrict__ L,
                                                float* __restrict__ O) {
  extern __shared__ char smem[];
  constexpr uint32_t QSH = 0, KS0 = 65536, KS1 = 98304,
                     VS0 = 131072, VS1 = 163840, PSH = 196608;
  __bf16* qsh  = (__bf16*)(smem + QSH);       // 64 x 512 (64 KB)
  __bf16* ksh0 = (__bf16*)(smem + KS0);       // 32 x 512 (32 KB)
  __bf16* ksh1 = (__bf16*)(smem + KS1);
  __bf16* vsh0 = (__bf16*)(smem + VS0);       // 512 x 32 (32 KB)
  __bf16* vsh1 = (__bf16*)(smem + VS1);
  __bf16* psh  = (__bf16*)(smem + PSH);       // 64 x 32  (4 KB)
  const uint32_t base = lds_base_addr(smem);

  const int b  = blockIdx.y;
  const int q0 = blockIdx.x * 64;
  const int wv = threadIdx.x >> 5;
  const int lane = threadIdx.x & 31;
  const int qg = wv >> 1;                   // 0..3
  const int og = wv & 1;                    // 0..1
  const int nn = lane & 15, hl = lane >> 4;

  auto stage = [&](int step, int buf) {
    copy_async_lin<2048>(base + (buf ? KS1 : KS0),
                         K + ((size_t)b * 2048 + step * 32) * 512);
    copy_async_rows64<512, 4096>(base + (buf ? VS1 : VS0),
                                 VT + (size_t)b * 512 * 2048 + (size_t)step * 32);
  };

  copy_async_lin<4096>(base + QSH, Q + ((size_t)b * 2048 + q0) * 512);
  stage(0, 0);

  v8f z = {};
  v8f acc[16];
#pragma unroll
  for (int t = 0; t < 16; ++t) acc[t] = z;

#pragma unroll 1
  for (int ks = 0; ks < 64; ++ks) {
    wait_async0();
    __syncthreads();
    if (ks + 1 < 64) stage(ks + 1, (ks + 1) & 1);
    const __bf16* kb = (ks & 1) ? ksh1 : ksh0;
    const __bf16* vb = (ks & 1) ? vsh1 : vsh0;

    // --- S tile: rows [q0+qg*16, +16), cols [ks*32+og*16, +16)
    v8f s = z;
#pragma unroll
    for (int kk = 0; kk < 16; ++kk) {
      v16bf a  = load_frag(qsh + (size_t)(qg * 16) * 512 + kk * 32, 512);
      v16bf bb = load_frag(kb  + (size_t)(og * 16) * 512 + kk * 32, 512);
      s = wmma_bf16(a, bb, s);
    }
    // --- P = exp(S - L[col])  -> bf16 in LDS
    const float Lv = L[(size_t)b * 2048 + ks * 32 + og * 16 + nn];
#pragma unroll
    for (int j = 0; j < 8; ++j) {
      const int m = qg * 16 + j + 8 * hl;
      psh[(size_t)m * 32 + og * 16 + nn] = (__bf16)__expf(s[j] - Lv);
    }
    __syncthreads();

    // --- out += P (16x32) x vt (32x16 per tile), K = 32 in one WMMA
    v16bf ap = load_frag(psh + (size_t)(qg * 16) * 32, 32);
#pragma unroll
    for (int t = 0; t < 16; ++t) {
      v16bf bv = load_frag(vb + (size_t)(og * 256 + t * 16) * 32, 32);
      acc[t] = wmma_bf16(ap, bv, acc[t]);
    }
  }

#pragma unroll
  for (int t = 0; t < 16; ++t)
#pragma unroll
    for (int j = 0; j < 8; ++j) {
      const int gq = q0 + qg * 16 + j + 8 * hl;
      const int go = og * 256 + t * 16 + nn;
      O[((size_t)b * 2048 + gq) * 512 + go] = acc[t][j];
    }
}

// ---------------------------------------------------------------------------
extern "C" void kernel_launch(void* const* d_in, const int* in_sizes, int n_in,
                              void* d_out, int out_size, void* d_ws, size_t ws_size,
                              hipStream_t stream) {
  const float* x  = (const float*)d_in[0];
  const float* Wq = (const float*)d_in[1];
  const float* Wk = (const float*)d_in[2];
  const float* Wv = (const float*)d_in[3];
  float* out = (float*)d_out;

  // workspace carve-out (~69 MB; harness scratch assumed sufficient)
  char* ws = (char*)d_ws;
  size_t off = 0;
  auto take = [&](size_t bytes) -> char* {
    char* p = ws + off;
    off = (off + bytes + 255) & ~(size_t)255;
    return p;
  };
  __bf16* xb  = (__bf16*)take((size_t)16384 * 512 * 2);
  __bf16* wqb = (__bf16*)take((size_t)512 * 512 * 2);
  __bf16* wkb = (__bf16*)take((size_t)512 * 512 * 2);
  __bf16* wvb = (__bf16*)take((size_t)512 * 512 * 2);
  __bf16* qb  = (__bf16*)take((size_t)16384 * 512 * 2);
  __bf16* kb  = (__bf16*)take((size_t)16384 * 512 * 2);
  __bf16* vt  = (__bf16*)take((size_t)8 * 512 * 2048 * 2);
  float*  Lg  = (float*)take((size_t)8 * 2048 * 4);
  (void)in_sizes; (void)n_in; (void)out_size; (void)ws_size;

  // 1) casts to bf16
  cast_f32_bf16<<<4096, 256, 0, stream>>>(x,  xb,  16384 * 512);
  cast_f32_bf16<<<256,  256, 0, stream>>>(Wq, wqb, 512 * 512);
  cast_f32_bf16<<<256,  256, 0, stream>>>(Wk, wkb, 512 * 512);
  cast_f32_bf16<<<256,  256, 0, stream>>>(Wv, wvb, 512 * 512);

  // 2) projections (q, k row-major; v transposed)  LDS: 2x(8K+8K)
  dim3 pg(4, 128);
  proj_gemm<<<pg, 256, 32768, stream>>>(xb, wqb, qb, 0);
  proj_gemm<<<pg, 256, 32768, stream>>>(xb, wkb, kb, 0);
  proj_gemm<<<pg, 256, 32768, stream>>>(xb, wvb, vt, 1);

  // 3) per-key-column logsumexp  (LDS: 128KB K panel + 2x16KB Q tiles)
  col_softmax_stats<<<dim3(16, 8), 256, 163840, stream>>>(qb, kb, Lg);

  // 4) output (LDS: 64KB Q + 2x32KB K + 2x32KB V^T + 4KB P)
  attn_out<<<dim3(32, 8), 256, 200704, stream>>>(qb, kb, vt, Lg, out);
}
